// SNNForecasterV6_89060441850417
// MI455X (gfx1250) — 16-bit and 8-bit matrix operand order repaired and checked by reading, compiled, not run on hardware
//
#include <hip/hip_runtime.h>

#define B_   16
#define T_   512
#define CIN  32
#define H_   256
#define HS_  1024
#define S_   2048
#define HZ_  96
#define BH   (B_*H_)
#define SBH  ((size_t)S_*B_*H_)
#define WPB  8

typedef float    v2f  __attribute__((ext_vector_type(2)));
typedef float    v8f  __attribute__((ext_vector_type(8)));
typedef int      v8i  __attribute__((ext_vector_type(8)));
typedef _Float16 v8h  __attribute__((ext_vector_type(8)));
typedef _Float16 v16h __attribute__((ext_vector_type(16)));

__device__ __forceinline__ unsigned lds_off(const void* p) {
  return (unsigned)(unsigned long long)p;
}
__device__ __forceinline__ void async_b128(unsigned dst, const void* src) {
  asm volatile("global_load_async_to_lds_b128 %0, %1, off"
               :: "v"(dst), "v"(src) : "memory");
}
__device__ __forceinline__ void async_b64(unsigned dst, const void* src) {
  asm volatile("global_load_async_to_lds_b64 %0, %1, off"
               :: "v"(dst), "v"(src) : "memory");
}
__device__ __forceinline__ void wait_async0() {
  asm volatile("s_wait_asynccnt 0x0" ::: "memory");
}

__global__ __launch_bounds__(256) void enc_gemm_bn_kernel(
    const float* __restrict__ x, const float* __restrict__ w,
    const float* __restrict__ bias,
    const float* __restrict__ g,  const float* __restrict__ bb,
    const float* __restrict__ rm, const float* __restrict__ rv,
    float* __restrict__ h_enc) {
  __shared__ float As[16*CIN];
  const int tid = threadIdx.x;
  const int wv = tid >> 5, lane = tid & 31;
  const int hl = lane >> 4, lm = lane & 15;
  const int row0 = blockIdx.x * 16;
  async_b64(lds_off(As) + tid*8, (const char*)(x + (size_t)row0*CIN) + tid*8);
  wait_async0();
  __syncthreads();
  const int col = blockIdx.y*128 + wv*16 + lm;
  const float* arow = As + lm*CIN;
  const float* brow = w + (size_t)col*CIN;
  v8f acc0 = {}, acc1 = {};
  #pragma unroll
  for (int k0 = 0; k0 < CIN; k0 += 8) {
    v2f a0 = *(const v2f*)(arow + k0 + 2*hl);
    v2f b0 = *(const v2f*)(brow + k0 + 2*hl);
    acc0 = __builtin_amdgcn_wmma_f32_16x16x4_f32(false, a0, false, b0, (short)0, acc0, false, false);
    v2f a1 = *(const v2f*)(arow + k0 + 4 + 2*hl);
    v2f b1 = *(const v2f*)(brow + k0 + 4 + 2*hl);
    acc1 = __builtin_amdgcn_wmma_f32_16x16x4_f32(false, a1, false, b1, (short)0, acc1, false, false);
  }
  const v8f acc = acc0 + acc1;
  const float sc = g[col] * rsqrtf(rv[col] + 1e-5f);
  const float sh = bb[col] + (bias[col] - rm[col]) * sc;
  const int ss = col >> 8, hh = col & (H_-1);
  #pragma unroll
  for (int r = 0; r < 8; ++r) {
    const int row = row0 + r + 8*hl;
    const int bb_ = row >> 9, tt = row & (T_-1);
    const int s = tt*4 + ss;
    h_enc[((size_t)s*B_ + bb_)*H_ + hh] = acc[r]*sc + sh;
  }
}

__global__ __launch_bounds__(256) void qkv_gemm_bn_kernel(
    const float* __restrict__ xa, const float* __restrict__ w,
    const float* __restrict__ bias,
    const float* __restrict__ g,  const float* __restrict__ bb,
    const float* __restrict__ rm, const float* __restrict__ rv,
    float* __restrict__ out) {
  __shared__ float As[16*H_];
  const int tid = threadIdx.x;
  const int wv = tid >> 5, lane = tid & 31;
  const int hl = lane >> 4, lm = lane & 15;
  const int row0 = blockIdx.x * 16;
  {
    const unsigned as0 = lds_off(As);
    const char* src = (const char*)(xa + (size_t)row0*H_);
    #pragma unroll
    for (int k = 0; k < 4; ++k)
      async_b128(as0 + k*4096 + tid*16, src + k*4096 + tid*16);
    wait_async0();
  }
  __syncthreads();
  const int col = blockIdx.y*128 + wv*16 + lm;
  const float* arow = As + lm*H_;
  const float* brow = w + (size_t)col*H_;
  v8f acc0 = {}, acc1 = {};
  #pragma unroll 4
  for (int k0 = 0; k0 < H_; k0 += 8) {
    v2f a0 = *(const v2f*)(arow + k0 + 2*hl);
    v2f b0 = *(const v2f*)(brow + k0 + 2*hl);
    acc0 = __builtin_amdgcn_wmma_f32_16x16x4_f32(false, a0, false, b0, (short)0, acc0, false, false);
    v2f a1 = *(const v2f*)(arow + k0 + 4 + 2*hl);
    v2f b1 = *(const v2f*)(brow + k0 + 4 + 2*hl);
    acc1 = __builtin_amdgcn_wmma_f32_16x16x4_f32(false, a1, false, b1, (short)0, acc1, false, false);
  }
  const v8f acc = acc0 + acc1;
  const float sc = g[col] * rsqrtf(rv[col] + 1e-5f);
  const float sh = bb[col] + (bias[col] - rm[col]) * sc;
  #pragma unroll
  for (int r = 0; r < 8; ++r) {
    const int row = row0 + r + 8*hl;
    out[(size_t)row*H_ + col] = acc[r]*sc + sh;
  }
}

__global__ void lif_enc_kernel(const float* __restrict__ h, const float* __restrict__ beta_p,
                               float* __restrict__ spk, float* __restrict__ ssum) {
  const int i = blockIdx.x * blockDim.x + threadIdx.x;
  const float beta = fminf(fmaxf(beta_p[0], 0.f), 1.f);
  float mem = 0.f, cnt = 0.f;
  for (int s = 0; s < S_; ++s) {
    __builtin_prefetch(h + (size_t)(s+4)*BH + i, 0, 1);
    const float inp = h[(size_t)s*BH + i];
    const float rst = mem > 1.f ? 1.f : 0.f;
    mem = beta*mem + inp - rst;
    const float sp = mem > 1.f ? 1.f : 0.f;
    spk[(size_t)s*BH + i] = sp;
    cnt += sp;
  }
  atomicAdd(ssum, cnt);
}

__global__ void lif23_kernel(const float* __restrict__ spk_enc,
                             const float* __restrict__ b2p, const float* __restrict__ b3p,
                             const float* __restrict__ g2,  const float* __restrict__ bb2,
                             const float* __restrict__ m2c, const float* __restrict__ v2c,
                             const float* __restrict__ g3,  const float* __restrict__ bb3,
                             const float* __restrict__ m3c, const float* __restrict__ v3c,
                             float* __restrict__ xa, float* __restrict__ ssum) {
  const int i = blockIdx.x * blockDim.x + threadIdx.x;
  const int b = i >> 8, hh = i & (H_-1);
  const float beta2 = fminf(fmaxf(b2p[0], 0.f), 1.f);
  const float beta3 = fminf(fmaxf(b3p[0], 0.f), 1.f);
  const float sc2 = g2[hh]*rsqrtf(v2c[hh]+1e-5f);
  const float sh2 = bb2[hh] - m2c[hh]*sc2;
  const float sc3 = g3[hh]*rsqrtf(v3c[hh]+1e-5f);
  const float sh3 = bb3[hh] - m3c[hh]*sc3;
  float m2 = 0.f, m3 = 0.f, cnt = 0.f;
  for (int s = 0; s < S_; ++s) {
    __builtin_prefetch(spk_enc + (size_t)(s+4)*BH + i, 0, 1);
    const float inp = spk_enc[(size_t)s*BH + i]*sc2 + sh2;
    const float r2 = m2 > 1.f ? 1.f : 0.f;
    m2 = beta2*m2 + inp - r2;
    const float s2 = m2 > 1.f ? 1.f : 0.f;
    const float r3 = m3 > 1.f ? 1.f : 0.f;
    m3 = beta3*m3 + (s2*sc3 + sh3) - r3;
    const float s3 = m3 > 1.f ? 1.f : 0.f;
    xa[((size_t)b*S_ + s)*H_ + hh] = m3;
    cnt += s2 + s3;
  }
  atomicAdd(ssum, cnt);
}

__global__ void lif_qkv_kernel(const float* __restrict__ pre, unsigned char* __restrict__ out8,
                               _Float16* __restrict__ outh, int as_half) {
  const int i = blockIdx.x * blockDim.x + threadIdx.x;
  const int b = i >> 8, hh = i & (H_-1);
  const float beta = 0.9f;
  float mem = 0.f;
  for (int s = 0; s < S_; ++s) {
    const float inp = pre[((size_t)b*S_ + s)*H_ + hh];
    const float rst = mem > 1.f ? 1.f : 0.f;
    mem = beta*mem + inp - rst;
    const float sp = mem > 1.f ? 1.f : 0.f;
    if (as_half) outh[((size_t)b*S_ + s)*H_ + hh] = (_Float16)sp;
    else         out8[((size_t)b*S_ + s)*H_ + hh] = (unsigned char)sp;
  }
}

__global__ __launch_bounds__(256) void attention_kernel(
    const unsigned char* __restrict__ Q8,
    const unsigned char* __restrict__ K8,
    const _Float16* __restrict__ Vh,
    float* __restrict__ feat) {
  __shared__ unsigned char Kl[32*H_];
  __shared__ _Float16      Vt[H_*32];
  __shared__ _Float16      Pt[WPB][16*32];
  const int tid = threadIdx.x;
  const int wv = tid >> 5, lane = tid & 31;
  const int hl = lane >> 4, lm = lane & 15;
  const int b  = blockIdx.y;
  const int q0 = (blockIdx.x * WPB + wv) * 16;
  const unsigned char* Qrow  = Q8 + ((size_t)b*S_ + q0 + lm) * H_;
  const unsigned char* Kbase = K8 + (size_t)b*S_*H_;
  const _Float16*      Vbase = Vh + (size_t)b*S_*H_;
  const unsigned kl0 = lds_off(Kl);

  v8i qa[4];
  #pragma unroll
  for (int c = 0; c < 4; ++c) {
    #pragma unroll
    for (int p = 0; p < 4; ++p) {
      const uint2 u = *(const uint2*)(Qrow + 64*c + 32*(p>>1) + 16*hl + 8*(p&1));
      qa[c][2*p]   = (int)u.x;
      qa[c][2*p+1] = (int)u.y;
    }
  }

  int imax[8];
  #pragma unroll
  for (int r = 0; r < 8; ++r) imax[r] = 0;
  for (int kb = 0; kb < S_; kb += 32) {
    {
      const char* src = (const char*)(Kbase + (size_t)kb*H_);
      async_b128(kl0 + tid*16,        src + tid*16);
      async_b128(kl0 + 4096 + tid*16, src + 4096 + tid*16);
      wait_async0();
    }
    __syncthreads();
    #pragma unroll
    for (int sub = 0; sub < 2; ++sub) {
      const unsigned char* Krow = Kl + (sub*16 + lm)*H_;
      v8i a0 = {}, a1 = {};
      #pragma unroll
      for (int c = 0; c < 2; ++c) {
        const uint4 u0 = *(const uint4*)(Krow + 64*c + 16*hl);
        const uint4 u1 = *(const uint4*)(Krow + 64*c + 32 + 16*hl);
        const uint4 u2 = *(const uint4*)(Krow + 128 + 64*c + 16*hl);
        const uint4 u3 = *(const uint4*)(Krow + 128 + 64*c + 32 + 16*hl);
        v8i kv0, kv1;
        kv0[0]=(int)u0.x; kv0[1]=(int)u0.y; kv0[2]=(int)u0.z; kv0[3]=(int)u0.w;
        kv0[4]=(int)u1.x; kv0[5]=(int)u1.y; kv0[6]=(int)u1.z; kv0[7]=(int)u1.w;
        kv1[0]=(int)u2.x; kv1[1]=(int)u2.y; kv1[2]=(int)u2.z; kv1[3]=(int)u2.w;
        kv1[4]=(int)u3.x; kv1[5]=(int)u3.y; kv1[6]=(int)u3.z; kv1[7]=(int)u3.w;
        a0 = __builtin_amdgcn_wmma_i32_16x16x64_iu8(false, qa[c],   false, kv0, a0, false, false);
        a1 = __builtin_amdgcn_wmma_i32_16x16x64_iu8(false, qa[c+2], false, kv1, a1, false, false);
      }
      const v8i acc = a0 + a1;
      #pragma unroll
      for (int r = 0; r < 8; ++r) imax[r] = acc[r] > imax[r] ? acc[r] : imax[r];
    }
    __syncthreads();
  }
  float rowmax[8];
  #pragma unroll
  for (int r = 0; r < 8; ++r) {
    int v = imax[r], t;
    t = __shfl_xor(v, 1); v = t > v ? t : v;
    t = __shfl_xor(v, 2); v = t > v ? t : v;
    t = __shfl_xor(v, 4); v = t > v ? t : v;
    t = __shfl_xor(v, 8); v = t > v ? t : v;
    rowmax[r] = (float)v * 0.0625f;
  }

  v8f O[16];
  #pragma unroll
  for (int nt = 0; nt < 16; ++nt) O[nt] = (v8f){};
  float lsum[8];
  #pragma unroll
  for (int r = 0; r < 8; ++r) lsum[r] = 0.f;

  for (int kb = 0; kb < S_; kb += 32) {
    {
      const char* src = (const char*)(Kbase + (size_t)kb*H_);
      async_b128(kl0 + tid*16,        src + tid*16);
      async_b128(kl0 + 4096 + tid*16, src + 4096 + tid*16);
    }
    {
      const _Float16* vs = Vbase + (size_t)kb*H_;
      #pragma unroll
      for (int c2 = 0; c2 < 4; ++c2) {
        const int chunk = tid + c2*256;
        const int key = chunk >> 5, h0 = (chunk & 31) * 8;
        const v8h v = *(const v8h*)(vs + (size_t)key*H_ + h0);
        #pragma unroll
        for (int e = 0; e < 8; ++e) Vt[(h0+e)*32 + key] = v[e];
      }
    }
    wait_async0();
    __syncthreads();
    _Float16* Ptw = Pt[wv];
    #pragma unroll
    for (int sub = 0; sub < 2; ++sub) {
      const unsigned char* Krow = Kl + (sub*16 + lm)*H_;
      v8i a0 = {}, a1 = {};
      #pragma unroll
      for (int c = 0; c < 2; ++c) {
        const uint4 u0 = *(const uint4*)(Krow + 64*c + 16*hl);
        const uint4 u1 = *(const uint4*)(Krow + 64*c + 32 + 16*hl);
        const uint4 u2 = *(const uint4*)(Krow + 128 + 64*c + 16*hl);
        const uint4 u3 = *(const uint4*)(Krow + 128 + 64*c + 32 + 16*hl);
        v8i kv0, kv1;
        kv0[0]=(int)u0.x; kv0[1]=(int)u0.y; kv0[2]=(int)u0.z; kv0[3]=(int)u0.w;
        kv0[4]=(int)u1.x; kv0[5]=(int)u1.y; kv0[6]=(int)u1.z; kv0[7]=(int)u1.w;
        kv1[0]=(int)u2.x; kv1[1]=(int)u2.y; kv1[2]=(int)u2.z; kv1[3]=(int)u2.w;
        kv1[4]=(int)u3.x; kv1[5]=(int)u3.y; kv1[6]=(int)u3.z; kv1[7]=(int)u3.w;
        a0 = __builtin_amdgcn_wmma_i32_16x16x64_iu8(false, qa[c],   false, kv0, a0, false, false);
        a1 = __builtin_amdgcn_wmma_i32_16x16x64_iu8(false, qa[c+2], false, kv1, a1, false, false);
      }
      const v8i acc = a0 + a1;
      #pragma unroll
      for (int r = 0; r < 8; ++r) {
        const float p = __expf((float)acc[r] * 0.0625f - rowmax[r]);
        lsum[r] += p;
        Ptw[(r + 8*hl)*32 + sub*16 + lm] = (_Float16)p;
      }
    }
    const v8h plo = *(const v8h*)(Ptw + lm*32 + 16*hl);
    const v8h phi = *(const v8h*)(Ptw + lm*32 + 16*hl + 8);
    v16h pa;
    #pragma unroll
    for (int j = 0; j < 8; ++j) { pa[j] = plo[j]; pa[8+j] = phi[j]; }
    #pragma unroll
    for (int nt = 0; nt < 16; ++nt) {
      const v16h vb = *(const v16h*)(Vt + (size_t)(nt*16 + lm)*32 + 16*hl);
      O[nt] = __builtin_amdgcn_wmma_f32_16x16x32_f16(false, pa, false, vb, (short)0, O[nt], false, false);
    }
    __syncthreads();
  }

  #pragma unroll
  for (int r = 0; r < 8; ++r) {
    float v = lsum[r];
    v += __shfl_xor(v, 1); v += __shfl_xor(v, 2);
    v += __shfl_xor(v, 4); v += __shfl_xor(v, 8);
    lsum[r] = 1.0f / v;
  }
  #pragma unroll
  for (int nt = 0; nt < 16; ++nt) {
    float a = 0.f;
    #pragma unroll
    for (int r = 0; r < 8; ++r) a += O[nt][r] * lsum[r];
    atomicAdd(&feat[b*H_ + nt*16 + lm], a * (1.0f/(float)S_));
  }
}

__global__ void head_kernel(const float* __restrict__ feat, const float* __restrict__ hw,
                            const float* __restrict__ hb, const float* __restrict__ ssum,
                            float* __restrict__ out) {
  const int i = blockIdx.x * blockDim.x + threadIdx.x;
  if (i < B_*HZ_) {
    const int b = i / HZ_, z = i % HZ_;
    const float* f  = feat + b*H_;
    const float* wz = hw + (size_t)z*H_;
    float a = 0.f;
    for (int h = 0; h < H_; ++h) a += f[h]*wz[h];
    out[i] = a + hb[z];
  }
  if (i == 0) out[B_*HZ_] = ssum[0];
}

__global__ void zero_kernel(float* __restrict__ p, int n) {
  const int i = blockIdx.x * blockDim.x + threadIdx.x;
  if (i < n) p[i] = 0.f;
}

extern "C" void kernel_launch(void* const* d_in, const int* in_sizes, int n_in,
                              void* d_out, int out_size, void* d_ws, size_t ws_size,
                              hipStream_t stream) {
  (void)in_sizes; (void)n_in; (void)out_size; (void)ws_size;
  const float* x        = (const float*)d_in[0];
  const float* conv_w   = (const float*)d_in[1];
  const float* conv_b   = (const float*)d_in[2];
  const float* bn1_g    = (const float*)d_in[3];
  const float* bn1_b    = (const float*)d_in[4];
  const float* bn1_m    = (const float*)d_in[5];
  const float* bn1_v    = (const float*)d_in[6];
  const float* beta_enc = (const float*)d_in[7];
  const float* bn2_g    = (const float*)d_in[8];
  const float* bn2_b    = (const float*)d_in[9];
  const float* bn2_m    = (const float*)d_in[10];
  const float* bn2_v    = (const float*)d_in[11];
  const float* beta2    = (const float*)d_in[12];
  const float* bn3_g    = (const float*)d_in[13];
  const float* bn3_b    = (const float*)d_in[14];
  const float* bn3_m    = (const float*)d_in[15];
  const float* bn3_v    = (const float*)d_in[16];
  const float* beta3    = (const float*)d_in[17];
  const float* wq       = (const float*)d_in[18];
  const float* bq       = (const float*)d_in[19];
  const float* bnq_g    = (const float*)d_in[20];
  const float* bnq_b    = (const float*)d_in[21];
  const float* bnq_m    = (const float*)d_in[22];
  const float* bnq_v    = (const float*)d_in[23];
  const float* wk       = (const float*)d_in[24];
  const float* bk       = (const float*)d_in[25];
  const float* bnk_g    = (const float*)d_in[26];
  const float* bnk_b    = (const float*)d_in[27];
  const float* bnk_m    = (const float*)d_in[28];
  const float* bnk_v    = (const float*)d_in[29];
  const float* wv       = (const float*)d_in[30];
  const float* bv       = (const float*)d_in[31];
  const float* bnv_g    = (const float*)d_in[32];
  const float* bnv_b    = (const float*)d_in[33];
  const float* bnv_m    = (const float*)d_in[34];
  const float* bnv_v    = (const float*)d_in[35];
  const float* head_w   = (const float*)d_in[36];
  const float* head_b   = (const float*)d_in[37];

  char* ws = (char*)d_ws;
  float* h_enc   = (float*)(ws);
  float* spk_enc = (float*)(ws + SBH*4);
  float* xa      = (float*)(ws + SBH*8);
  float* vpre    = (float*)(ws + SBH*12);
  float* qpre    = h_enc;
  float* kpre    = spk_enc;
  unsigned char* Q8 = (unsigned char*)(ws + SBH*16);
  unsigned char* K8 = Q8 + SBH;
  _Float16* Vh   = (_Float16*)(K8 + SBH);
  float* feat    = (float*)((char*)Vh + SBH*2);
  float* ssum    = feat + BH;

  zero_kernel<<<(BH + 1 + 255)/256, 256, 0, stream>>>(feat, BH + 1);

  enc_gemm_bn_kernel<<<dim3((B_*T_)/16, HS_/128), 256, 0, stream>>>(
      x, conv_w, conv_b, bn1_g, bn1_b, bn1_m, bn1_v, h_enc);

  lif_enc_kernel<<<BH/256, 256, 0, stream>>>(h_enc, beta_enc, spk_enc, ssum);

  lif23_kernel<<<BH/256, 256, 0, stream>>>(spk_enc, beta2, beta3,
      bn2_g, bn2_b, bn2_m, bn2_v, bn3_g, bn3_b, bn3_m, bn3_v, xa, ssum);

  const dim3 ggrid((B_*S_)/16, H_/128);
  qkv_gemm_bn_kernel<<<ggrid, 256, 0, stream>>>(xa, wq, bq, bnq_g, bnq_b, bnq_m, bnq_v, qpre);
  qkv_gemm_bn_kernel<<<ggrid, 256, 0, stream>>>(xa, wk, bk, bnk_g, bnk_b, bnk_m, bnk_v, kpre);
  qkv_gemm_bn_kernel<<<ggrid, 256, 0, stream>>>(xa, wv, bv, bnv_g, bnv_b, bnv_m, bnv_v, vpre);

  lif_qkv_kernel<<<BH/256, 256, 0, stream>>>(qpre, Q8, nullptr, 0);
  lif_qkv_kernel<<<BH/256, 256, 0, stream>>>(kpre, K8, nullptr, 0);
  lif_qkv_kernel<<<BH/256, 256, 0, stream>>>(vpre, nullptr, Vh, 1);

  attention_kernel<<<dim3(S_/(16*WPB), B_), 256, 0, stream>>>(Q8, K8, Vh, feat);

  head_kernel<<<(B_*HZ_ + 255)/256, 256, 0, stream>>>(feat, head_w, head_b, ssum, (float*)d_out);
}
